// TrafficGNN_12893491822881
// MI455X (gfx1250) — matrix operand order repaired and checked by reading, not compiled, not run on hardware
//
#include <hip/hip_runtime.h>

// ---------------------------------------------------------------------------
// GCN on MI455X (gfx1250, wave32).
//  * GEMMs: v_wmma_f32_16x16x32_bf16, A converted f32->bf16 in-register,
//    W staged once per block into LDS (bf16, B-fragment-swizzled) so the
//    inner loop reads each 16x16 B fragment with two ds_load_b128.
//  * Edge aggregation: coalesced b128 gathers + global_atomic_add_f32.
//    xw/agg (51.2 MB each) are L2-resident (192 MB), so the random
//    scatter/gather never touches HBM; HBM traffic ~200 MB => ~10 us floor
//    at 23.3 TB/s, the rest is L2 atomic throughput.
// ---------------------------------------------------------------------------

typedef __attribute__((ext_vector_type(16))) __bf16 v16bf;
typedef __attribute__((ext_vector_type(8)))  __bf16 v8bf;
typedef __attribute__((ext_vector_type(8)))  float  v8f;

static __device__ __forceinline__ v16bf pack16(const float4& a0, const float4& a1,
                                               const float4& a2, const float4& a3) {
  v16bf r;
  r[0]  = (__bf16)a0.x; r[1]  = (__bf16)a0.y; r[2]  = (__bf16)a0.z; r[3]  = (__bf16)a0.w;
  r[4]  = (__bf16)a1.x; r[5]  = (__bf16)a1.y; r[6]  = (__bf16)a1.z; r[7]  = (__bf16)a1.w;
  r[8]  = (__bf16)a2.x; r[9]  = (__bf16)a2.y; r[10] = (__bf16)a2.z; r[11] = (__bf16)a2.w;
  r[12] = (__bf16)a3.x; r[13] = (__bf16)a3.y; r[14] = (__bf16)a3.z; r[15] = (__bf16)a3.w;
  return r;
}

// Stage W[K,N] (f32, row-major) into LDS as bf16, swizzled so that the 16
// K-values of column c in K-block kb are contiguous:
//   lw[((kb*N + c) << 4) + (k & 15)] = bf16(W[k][c])
// Global reads are coalesced (c fastest); two consecutive K rows per thread
// pack into adjacent bf16 slots.
static __device__ __forceinline__ void fill_w_lds(__bf16* lw, const float* __restrict__ W,
                                                  int K, int N) {
  const int total = (K >> 1) * N;
  for (int idx = threadIdx.x; idx < total; idx += blockDim.x) {
    const int kp = idx / N;
    const int c  = idx - kp * N;
    const int k  = kp << 1;
    const float lo = W[(size_t)k * N + c];
    const float hi = W[(size_t)(k + 1) * N + c];
    __bf16* g = lw + ((((k >> 4) * N + c) << 4) + (k & 15));
    g[0] = (__bf16)lo;
    g[1] = (__bf16)hi;
  }
}

// One wave accumulates a 16x64 tile (4 n-subtiles) of A[M,K] @ W[K,N].
// Fragment layouts used here (wave32, contiguous k order, same for A and B):
//   A 16x32 bf16: lane<16 -> K base 0, lane>=16 -> +16; elems 0..15 = K+0..15.
//   B 32x16 bf16: lane<16 -> K rows 0..15, lane>=16 -> +16; N = lane&15.
static __device__ __forceinline__ void wave_gemm4_lds(const float* __restrict__ A,
                                                      const __bf16* __restrict__ LW,
                                                      int K, int N, int row, int lane,
                                                      int n0, v8f acc[4]) {
  const int hiA = (lane >> 4) << 4;
  const int hiB = (lane >> 4) << 4;
  const int col = lane & 15;
  for (int k0 = 0; k0 < K; k0 += 32) {
    const float* ap = A + (size_t)row * K + (k0 + hiA);
    float4 a0 = *(const float4*)(ap + 0);
    float4 a1 = *(const float4*)(ap + 4);
    float4 a2 = *(const float4*)(ap + 8);
    float4 a3 = *(const float4*)(ap + 12);
    v16bf a = pack16(a0, a1, a2, a3);
#pragma unroll
    for (int t = 0; t < 4; ++t) {
      const __bf16* bp = LW + ((((k0 + hiB) >> 4) * N + (n0 + (t << 4) + col)) << 4);
      const v8bf blo = ((const v8bf*)bp)[0];      // ds_load_b128
      const v8bf bhi = ((const v8bf*)bp)[1];      // ds_load_b128
      const v16bf b = __builtin_shufflevector(blo, bhi,
          0, 1, 2, 3, 4, 5, 6, 7, 8, 9, 10, 11, 12, 13, 14, 15);
      acc[t] = __builtin_amdgcn_wmma_f32_16x16x32_bf16(
          false, a, false, b, (short)0, acc[t], false, false);
    }
  }
}

// C[M,N] = A[M,K] @ W[K,N].  8 waves/block; wave -> 16x64 tile.
// Dynamic LDS: K*N*2 bytes.
__global__ void __launch_bounds__(256) k_gemm(const float* __restrict__ A,
                                              const float* __restrict__ W,
                                              float* __restrict__ C,
                                              int M, int K, int N) {
  extern __shared__ __bf16 lw[];
  fill_w_lds(lw, W, K, N);
  __syncthreads();

  const int lane = threadIdx.x & 31;
  const int wid  = threadIdx.x >> 5;
  const int wps  = N >> 6;                       // waves per 16-row strip
  const int gw   = blockIdx.x * 8 + wid;
  const int strip = gw / wps;
  if (strip * 16 >= M) return;
  const int n0  = (gw % wps) << 6;
  const int m0  = strip << 4;
  const int row = m0 + (lane & 15);

  v8f acc[4] = {};
  wave_gemm4_lds(A, lw, K, N, row, lane, n0, acc);

  const int srow = m0 + ((lane >> 4) << 3);
  const int col  = lane & 15;
#pragma unroll
  for (int t = 0; t < 4; ++t)
#pragma unroll
    for (int r = 0; r < 8; ++r)
      C[(size_t)(srow + r) * N + (n0 + (t << 4) + col)] = acc[t][r];
}

// out = H @ Wf + X @ Wsk + bf + bsk   (N == 64: one wave covers all columns)
// Dynamic LDS: (Kh + Kx) * N * 2 bytes.
__global__ void __launch_bounds__(256) k_final(const float* __restrict__ H,
                                               const float* __restrict__ Wf,
                                               const float* __restrict__ X,
                                               const float* __restrict__ Wsk,
                                               const float* __restrict__ bf,
                                               const float* __restrict__ bsk,
                                               float* __restrict__ out,
                                               int M, int Kh, int Kx, int N) {
  extern __shared__ __bf16 lw[];
  __bf16* lwf  = lw;
  __bf16* lwsk = lw + (size_t)Kh * N;
  fill_w_lds(lwf, Wf, Kh, N);
  fill_w_lds(lwsk, Wsk, Kx, N);
  __syncthreads();

  const int lane  = threadIdx.x & 31;
  const int wid   = threadIdx.x >> 5;
  const int strip = blockIdx.x * 8 + wid;
  if (strip * 16 >= M) return;
  const int m0  = strip << 4;
  const int row = m0 + (lane & 15);

  v8f acc[4] = {};
  wave_gemm4_lds(H, lwf, Kh, N, row, lane, 0, acc);
  wave_gemm4_lds(X, lwsk, Kx, N, row, lane, 0, acc);

  const int srow = m0 + ((lane >> 4) << 3);
  const int col  = lane & 15;
#pragma unroll
  for (int t = 0; t < 4; ++t) {
    const float bb = bf[(t << 4) + col] + bsk[(t << 4) + col];
#pragma unroll
    for (int r = 0; r < 8; ++r)
      out[(size_t)(srow + r) * N + ((t << 4) + col)] = acc[t][r] + bb;
  }
}

__global__ void k_zero_i32(int* __restrict__ p, int n) {
  int i = blockIdx.x * blockDim.x + threadIdx.x;
  if (i < n) p[i] = 0;
}

__global__ void k_count_deg(int* __restrict__ cnt, const int* __restrict__ dst, int E) {
  int i = blockIdx.x * blockDim.x + threadIdx.x;
  if (i < E) atomicAdd(&cnt[dst[i]], 1);
}

__global__ void k_dinv(const int* __restrict__ cnt, float* __restrict__ dinv, int n) {
  int i = blockIdx.x * blockDim.x + threadIdx.x;
  if (i < n) dinv[i] = rsqrtf((float)cnt[i] + 1.0f);   // +1 self-loop
}

// agg = xw * dinv[v]^2  (self-loop term; initializes agg, no separate zeroing)
__global__ void k_selfloop(const float* __restrict__ xw, const float* __restrict__ dinv,
                           float* __restrict__ agg, int n, int F) {
  int i = blockIdx.x * blockDim.x + threadIdx.x;        // one float4 per thread
  int total = n * (F >> 2);
  if (i >= total) return;
  int base = i << 2;
  int v = base / F;
  float s = dinv[v] * dinv[v];
  float4 w = *(const float4*)(xw + base);
  float4 o; o.x = w.x * s; o.y = w.y * s; o.z = w.z * s; o.w = w.w * s;
  *(float4*)(agg + base) = o;
}

// agg[dst] += xw[src] * dinv[src]*dinv[dst]   (4 features/thread, b128 gather)
__global__ void k_scatter(const float* __restrict__ xw, const float* __restrict__ dinv,
                          const int* __restrict__ src, const int* __restrict__ dst,
                          float* __restrict__ agg, int E, int F) {
  unsigned i = blockIdx.x * blockDim.x + threadIdx.x;
  const int per = F >> 2;                      // float4 groups per edge
  int e  = (int)(i / per);
  if (e >= E) return;
  int fg = (int)(i % per);
  int s = src[e], d = dst[e];
  float norm = dinv[s] * dinv[d];
  const float4 v = *(const float4*)(xw + (size_t)s * F + (fg << 2));
  float* o = agg + (size_t)d * F + (fg << 2);
  unsafeAtomicAdd(o + 0, v.x * norm);
  unsafeAtomicAdd(o + 1, v.y * norm);
  unsafeAtomicAdd(o + 2, v.z * norm);
  unsafeAtomicAdd(o + 3, v.w * norm);
}

// agg = relu(agg + b)   (in-place; produces the layer activation)
__global__ void k_bias_relu(float* __restrict__ agg, const float* __restrict__ b,
                            int n, int F) {
  int i = blockIdx.x * blockDim.x + threadIdx.x;
  int total = n * (F >> 2);
  if (i >= total) return;
  int base = i << 2;
  int f = base % F;
  float4 bv = *(const float4*)(b + f);
  float4 v  = *(const float4*)(agg + base);
  v.x = fmaxf(v.x + bv.x, 0.0f);
  v.y = fmaxf(v.y + bv.y, 0.0f);
  v.z = fmaxf(v.z + bv.z, 0.0f);
  v.w = fmaxf(v.w + bv.w, 0.0f);
  *(float4*)(agg + base) = v;
}

static inline unsigned cdiv(size_t a, size_t b) { return (unsigned)((a + b - 1) / b); }

extern "C" void kernel_launch(void* const* d_in, const int* in_sizes, int n_in,
                              void* d_out, int out_size, void* d_ws, size_t ws_size,
                              hipStream_t stream) {
  const float* x    = (const float*)d_in[0];
  const int*   ei   = (const int*)  d_in[1];
  const float* W1   = (const float*)d_in[2];
  const float* b1   = (const float*)d_in[3];
  const float* W2   = (const float*)d_in[4];
  const float* b2   = (const float*)d_in[5];
  const float* Wf   = (const float*)d_in[6];
  const float* bff  = (const float*)d_in[7];
  const float* Wsk  = (const float*)d_in[8];
  const float* bsk  = (const float*)d_in[9];

  const int HID  = in_sizes[3];                  // 256
  const int OUT  = in_sizes[7];                  // 64
  const int IND  = in_sizes[2] / HID;            // 128
  const int n    = in_sizes[0] / IND;            // 50000
  const int E    = in_sizes[1] / 2;              // 800000
  const int* src = ei;
  const int* dst = ei + E;

  // workspace: [cnt:int n][dinv:f32 n][xw:f32 n*HID][agg:f32 n*HID]  (~103 MB)
  const size_t nA = ((size_t)n + 255) & ~(size_t)255;
  int*   cnt  = (int*)d_ws;
  float* dinv = (float*)d_ws + nA;
  float* xw   = (float*)d_ws + 2 * nA;
  float* agg  = xw + (size_t)n * HID;

  const int T = 256;
  const unsigned gN  = cdiv(n, T);
  const unsigned gE  = cdiv(E, T);
  const unsigned gNF = cdiv((size_t)n * (HID >> 2), T);
  const unsigned gSc = cdiv((size_t)E * (HID >> 2), T);
  const int strips   = (n + 15) / 16;

  const size_t lds1 = (size_t)IND * HID * 2;            // 64 KB
  const size_t lds2 = (size_t)HID * HID * 2;            // 128 KB
  const size_t ldsF = (size_t)(HID + IND) * OUT * 2;    // 48 KB

  // --- degrees / normalization ---
  k_zero_i32 <<<gN, T, 0, stream>>>(cnt, n);
  k_count_deg<<<gE, T, 0, stream>>>(cnt, dst, E);
  k_dinv     <<<gN, T, 0, stream>>>(cnt, dinv, n);

  // --- layer 1: xw = x @ W1 ; aggregate ; h1 = relu(agg + b1) (in agg) ---
  k_gemm<<<cdiv((size_t)strips * (HID >> 6), 8), T, lds1, stream>>>(x, W1, xw, n, IND, HID);
  k_selfloop <<<gNF, T, 0, stream>>>(xw, dinv, agg, n, HID);
  k_scatter  <<<gSc, T, 0, stream>>>(xw, dinv, src, dst, agg, E, HID);
  k_bias_relu<<<gNF, T, 0, stream>>>(agg, b1, n, HID);

  // --- layer 2: xw = h1 @ W2 ; aggregate ; h2 = relu(agg + b2) (in agg) ---
  k_gemm<<<cdiv((size_t)strips * (HID >> 6), 8), T, lds2, stream>>>(agg, W2, xw, n, HID, HID);
  k_selfloop <<<gNF, T, 0, stream>>>(xw, dinv, agg, n, HID);
  k_scatter  <<<gSc, T, 0, stream>>>(xw, dinv, src, dst, agg, E, HID);
  k_bias_relu<<<gNF, T, 0, stream>>>(agg, b2, n, HID);

  // --- head: out = h2 @ Wf + x @ Wsk + bf + bsk ---
  k_final<<<cdiv(strips, 8), T, ldsF, stream>>>(agg, Wf, x, Wsk, bff, bsk,
                                                (float*)d_out, n, HID, IND, OUT);
}
